// Graph_CNN_ortega_87548613361704
// MI455X (gfx1250) — matrix operand order repaired and checked by reading, not compiled, not run on hardware
//
#include <hip/hip_runtime.h>
#include <hip/hip_bf16.h>

typedef unsigned short u16;
typedef unsigned int u32;
typedef unsigned long long u64;

typedef __attribute__((ext_vector_type(16))) __bf16 v16bf;
typedef __attribute__((ext_vector_type(8)))  float  v8f;
typedef __attribute__((ext_vector_type(4)))  unsigned int v4u;
typedef __attribute__((ext_vector_type(8)))  int v8i;
typedef __attribute__((ext_vector_type(4)))  int v4i;

union Frag { v16bf v; uint4 q[2]; };

#define BM 128
#define BN 128
#define BK 32
#define LDS_K 40   // BK + 8 pad: keeps 16B alignment, breaks bank conflicts

enum { EP_F32 = 0, EP_F32_ACC = 1, EP_BF16 = 2, EP_BF16_BIAS_RELU = 3, EP_BF16_BIAS_SCALE = 4 };

// dynamic LDS arena layout (bytes): A buf0, A buf1, B buf0, B buf1
#define A_TILE_ELEMS (BM * LDS_K)
#define B_TILE_ELEMS (BN * LDS_K)
#define GEMM_LDS_BYTES ((2 * A_TILE_ELEMS + 2 * B_TILE_ELEMS) * 2)

__device__ __forceinline__ u16 f2bf(float f) {
    u32 u = __float_as_uint(f);
    u32 r = (u + 0x7FFFu + ((u >> 16) & 1u)) >> 16;   // round-to-nearest-even
    return (u16)r;
}

// ---------------------------------------------------------------------------
// TDM descriptor issue: DMA a [BM x BK] bf16 tile (row stride = lda elems)
// from global memory into LDS at byte offset ldsByte, padding each 64B row
// to 80B (matches LDS_K=40 pitch). 2-D tile -> groups 2/3 are zero.
// ---------------------------------------------------------------------------
__device__ __forceinline__ void tdm_load_tileA(const u16* gptr, u32 ldsByte,
                                               int Mtot, int Ktot, int lda) {
    const u64 ga = (u64)(size_t)gptr;
    v4u g0;
    g0[0] = 1u;                                     // count=1 (valid user D#)
    g0[1] = ldsByte;                                // lds_addr (bytes)
    g0[2] = (u32)(ga & 0xFFFFFFFFu);                // global_addr[31:0]
    g0[3] = (u32)((ga >> 32) & 0x01FFFFFFu)         // global_addr[56:32]
          | (2u << 30);                             // type = 2 ("image")
    v8i g1;
    // data_size=1 (2B), pad_enable=1, pad_interval=3 (16 DW), pad_amount=3 (4 DW)
    g1[0] = (int)((1u << 16) | (1u << 20) | (3u << 22) | (3u << 25));
    g1[1] = (int)(((u32)Ktot & 0xFFFFu) << 16);     // tensor_dim0 lo16 @ [31:16]
    g1[2] = (int)((((u32)Ktot >> 16) & 0xFFFFu)     // tensor_dim0 hi16
          | (((u32)Mtot & 0xFFFFu) << 16));         // tensor_dim1 lo16
    g1[3] = (int)((((u32)Mtot >> 16) & 0xFFFFu)     // tensor_dim1 hi16
          | ((u32)BK << 16));                       // tile_dim0 = 32 elems
    g1[4] = (int)BM;                                // tile_dim1 = 128 rows (tile_dim2=0)
    g1[5] = (int)lda;                               // tensor_dim0_stride lo32 (elems)
    g1[6] = 0;                                      // stride hi16 | dim1_stride lo16
    g1[7] = 0;
    const v4i z4 = (v4i){0, 0, 0, 0};
#if defined(__clang_major__) && (__clang_major__ >= 23)
    const v8i z8 = (v8i){0, 0, 0, 0, 0, 0, 0, 0};
    __builtin_amdgcn_tensor_load_to_lds(g0, g1, z4, z4, z8, 0);
#else
    __builtin_amdgcn_tensor_load_to_lds(g0, g1, z4, z4, 0);
#endif
}

// ---------------------------------------------------------------------------
// bf16 WMMA GEMM: C[M,Ncols] = A[M,K] * B[K,Ncols], f32 accumulate.
// A tile: CDNA5 Tensor Data Mover DMA global->LDS (TENSORcnt-tracked,
//         hardware row padding), issued by wave 0 only.
// B tile: VGPR-staged transpose into LDS so fragments are contiguous b128s.
// Fused epilogues select f32 store/accum or bf16 (+bias/relu/scale) output.
// ---------------------------------------------------------------------------
__global__ __launch_bounds__(256) void
gemm_bf16_wmma(const u16* __restrict__ A, long long sA, int lda,
               const u16* __restrict__ Bm, long long sB, int ldb,
               void* __restrict__ Cout, long long sC, int ldc,
               const float* __restrict__ bias,
               const float* __restrict__ scaleP,
               int M, int Ncols, int K, int mode)
{
    extern __shared__ u16 smem[];   // dynamic => starts at LDS byte offset 0

    const int tid   = threadIdx.x;
    const int lane  = tid & 31;
    const int wave  = tid >> 5;
    const int waveM = wave >> 2;        // 0..1  (64 rows each)
    const int waveN = wave & 3;         // 0..3  (32 cols each)
    const int lhalf = lane & 15;
    const int hi    = lane >> 4;

    const int bMo = blockIdx.x * BM;
    const int bNo = blockIdx.y * BN;
    const int z   = blockIdx.z;

    const u16* Ab = A  + (size_t)z * (size_t)sA;
    const u16* Bb = Bm + (size_t)z * (size_t)sB;

    // B fill (with transpose): thread covers row-pair 2*b_rp/2*b_rp+1, cols b_cg*8..+7.
    const int b_rp = tid >> 4;               // 0..15
    const int b_cg = tid & 15;               // 0..15
    const bool bIn = (bNo + b_cg * 8) < Ncols;   // Ncols is a multiple of 8

    uint4 rb0, rb1;
    const int nk = K / BK;

    auto fillA_tdm = [&](int kt, int buf) {
        // one DMA descriptor covers the whole 128x32 tile incl. LDS row padding
        if (wave == 0) {
            const u16* g = Ab + (size_t)bMo * lda + kt * BK;
            tdm_load_tileA(g, (u32)(buf * A_TILE_ELEMS * 2), M, K, lda);
        }
    };
    auto fillB_regs = [&](int kt) {
        const int k0 = kt * BK;
        if (bIn) {
            rb0 = *(const uint4*)(Bb + (size_t)(k0 + 2 * b_rp)     * ldb + bNo + b_cg * 8);
            rb1 = *(const uint4*)(Bb + (size_t)(k0 + 2 * b_rp + 1) * ldb + bNo + b_cg * 8);
        } else {
            rb0 = make_uint4(0u, 0u, 0u, 0u);
            rb1 = make_uint4(0u, 0u, 0u, 0u);
        }
    };
    auto storeB_lds = [&](int buf) {
        u16* Bs = smem + 2 * A_TILE_ELEMS + buf * B_TILE_ELEMS;
        const u16* p0 = (const u16*)&rb0;
        const u16* p1 = (const u16*)&rb1;
#pragma unroll
        for (int i = 0; i < 8; ++i) {
            u32 pk = (u32)p0[i] | ((u32)p1[i] << 16);   // k=2rp, k=2rp+1 adjacent
            *(u32*)&Bs[(b_cg * 8 + i) * LDS_K + 2 * b_rp] = pk;
        }
    };

    v8f acc[4][2];
#pragma unroll
    for (int i = 0; i < 4; ++i)
#pragma unroll
        for (int j = 0; j < 2; ++j)
            acc[i][j] = (v8f){0.f, 0.f, 0.f, 0.f, 0.f, 0.f, 0.f, 0.f};

    fillA_tdm(0, 0);
    fillB_regs(0);
    storeB_lds(0);
    if (wave == 0) __builtin_amdgcn_s_wait_tensorcnt(0);
    __syncthreads();

    for (int kt = 0; kt < nk; ++kt) {
        const int cur = kt & 1;
        const bool pf = (kt + 1 < nk);
        if (pf) {
            fillA_tdm(kt + 1, 1 - cur);   // TDM DMA overlaps the WMMAs below
            fillB_regs(kt + 1);
        }

        const u16* As = smem + cur * A_TILE_ELEMS;
        const u16* Bs = smem + 2 * A_TILE_ELEMS + cur * B_TILE_ELEMS;

        Frag fa[4], fb[2];
#pragma unroll
        for (int i = 0; i < 4; ++i) {
            // A 16x32 frag: lane = row (M), halves hold K {hi*8..+7} and {16+hi*8..+7}
            const u16* p = As + (waveM * 64 + i * 16 + lhalf) * LDS_K;
            fa[i].q[0] = *(const uint4*)(p + hi * 8);
            fa[i].q[1] = *(const uint4*)(p + 16 + hi * 8);
        }
#pragma unroll
        for (int j = 0; j < 2; ++j) {
            // B 32x16 frag from transposed LDS: lane = col (N), halves hold K {hi*8..+7} and {16+hi*8..+7}
            const u16* p = Bs + (waveN * 32 + j * 16 + lhalf) * LDS_K;
            fb[j].q[0] = *(const uint4*)(p + hi * 8);
            fb[j].q[1] = *(const uint4*)(p + 16 + hi * 8);
        }
#pragma unroll
        for (int i = 0; i < 4; ++i)
#pragma unroll
            for (int j = 0; j < 2; ++j)
                acc[i][j] = __builtin_amdgcn_wmma_f32_16x16x32_bf16(
                    false, fa[i].v, false, fb[j].v, (short)0, acc[i][j], false, false);

        if (pf) {
            storeB_lds(1 - cur);
            if (wave == 0) __builtin_amdgcn_s_wait_tensorcnt(0);  // our DMA landed
        }
        __syncthreads();
    }

    // Epilogue. C tile layout: VGPR r -> row (hi*8 + r), col = lane&15.
    const int colBase = bNo + waveN * 32 + lhalf;
    const int rowBase = bMo + waveM * 64 + hi * 8;

    if (mode <= EP_F32_ACC) {
        float* C = (float*)Cout + (size_t)z * (size_t)sC;
#pragma unroll
        for (int i = 0; i < 4; ++i)
#pragma unroll
            for (int j = 0; j < 2; ++j) {
                int col = colBase + j * 16;
                if (col < Ncols) {
#pragma unroll
                    for (int r = 0; r < 8; ++r) {
                        size_t idx = (size_t)(rowBase + i * 16 + r) * ldc + col;
                        float v = acc[i][j][r];
                        if (mode == EP_F32_ACC) v += C[idx];
                        C[idx] = v;
                    }
                }
            }
    } else {
        u16* C = (u16*)Cout + (size_t)z * (size_t)sC;
        float sc = (mode == EP_BF16_BIAS_SCALE) ? scaleP[0] : 1.0f;
#pragma unroll
        for (int i = 0; i < 4; ++i)
#pragma unroll
            for (int j = 0; j < 2; ++j) {
                int col = colBase + j * 16;
                if (col < Ncols) {
                    float bv = (mode >= EP_BF16_BIAS_RELU) ? bias[col] : 0.0f;
#pragma unroll
                    for (int r = 0; r < 8; ++r) {
                        float v = acc[i][j][r] + bv;
                        if (mode == EP_BF16_BIAS_RELU) v = v > 0.f ? v : 0.f;
                        v *= sc;
                        C[(size_t)(rowBase + i * 16 + r) * ldc + col] = f2bf(v);
                    }
                }
            }
    }
}

// ---------------------------------------------------------------------------
// Support kernels
// ---------------------------------------------------------------------------
__global__ void k_conv_bf16(const float* __restrict__ in, u16* __restrict__ out, long long n) {
    long long i = (long long)blockIdx.x * blockDim.x + threadIdx.x;
    long long st = (long long)gridDim.x * blockDim.x;
    for (; i < n; i += st) out[i] = f2bf(in[i]);
}

__global__ void k_relu_conv_bf16(const float* __restrict__ in, u16* __restrict__ out, long long n) {
    long long i = (long long)blockIdx.x * blockDim.x + threadIdx.x;
    long long st = (long long)gridDim.x * blockDim.x;
    for (; i < n; i += st) { float v = in[i]; out[i] = f2bf(v > 0.f ? v : 0.f); }
}

__global__ void k_zero_f32(float* __restrict__ p, long long n) {
    long long i = (long long)blockIdx.x * blockDim.x + threadIdx.x;
    long long st = (long long)gridDim.x * blockDim.x;
    for (; i < n; i += st) p[i] = 0.f;
}

// Convert U to bf16 and also write U^T (bf16), LDS-tiled 32x32. grid.z = branch.
__global__ void k_convT(const float* __restrict__ U, u16* __restrict__ Ub,
                        u16* __restrict__ UTb, int n) {
    __shared__ float tile[32][33];
    const int k  = blockIdx.z;
    const float* src = U + (size_t)k * n * n;
    const int bx = blockIdx.x * 32;   // source col base
    const int by = blockIdx.y * 32;   // source row base
#pragma unroll
    for (int r = 0; r < 4; ++r) {
        int row = by + threadIdx.y * 4 + r;
        int col = bx + threadIdx.x;
        float v = src[(size_t)row * n + col];
        tile[threadIdx.y * 4 + r][threadIdx.x] = v;
        Ub[(size_t)k * n * n + (size_t)row * n + col] = f2bf(v);
    }
    __syncthreads();
#pragma unroll
    for (int r = 0; r < 4; ++r) {
        int row = bx + threadIdx.y * 4 + r;   // output row = source col
        int col = by + threadIdx.x;           // output col = source row
        UTb[(size_t)k * n * n + (size_t)row * n + col] = f2bf(tile[threadIdx.x][threadIdx.y * 4 + r]);
    }
}

__global__ void k_softmax3(const float* __restrict__ bw, float* __restrict__ wsb, int L) {
    int l = threadIdx.x;
    if (l < L) {
        float a = bw[l * 3 + 0], b = bw[l * 3 + 1], c = bw[l * 3 + 2];
        float m = fmaxf(a, fmaxf(b, c));
        float ea = __expf(a - m), eb = __expf(b - m), ec = __expf(c - m);
        float inv = 1.f / (ea + eb + ec);
        wsb[l * 3 + 0] = ea * inv;
        wsb[l * 3 + 1] = eb * inv;
        wsb[l * 3 + 2] = ec * inv;
    }
}

// Mean-pool over nodes (with final ReLU) + classifier head. One block per batch.
__global__ __launch_bounds__(128) void
k_pool_head(const float* __restrict__ accum, const float* __restrict__ Wc1,
            const float* __restrict__ bc1, const float* __restrict__ alpha,
            const float* __restrict__ Wc2, const float* __restrict__ bc2,
            float* __restrict__ out) {
    const int b = blockIdx.x;
    const int d = threadIdx.x;               // 0..127
    const float* base = accum + (size_t)b * 1024 * 128;
    float s = 0.f;
    for (int nn = 0; nn < 1024; ++nn) {
        float v = base[nn * 128 + d];
        s += (v > 0.f ? v : 0.f);
    }
    s *= (1.0f / 1024.0f);

    __shared__ float pooled[128];
    __shared__ float zbuf[128];
    pooled[d] = s;
    __syncthreads();

    float zv = bc1[d];
    for (int i = 0; i < 128; ++i) zv += pooled[i] * Wc1[i * 128 + d];
    zv = (zv > 0.f) ? zv : alpha[d] * zv;    // PReLU
    zbuf[d] = zv;
    __syncthreads();

    if (d < 4) {
        float o = bc2[d];
        for (int h = 0; h < 128; ++h) o += zbuf[h] * Wc2[h * 4 + d];
        out[b * 4 + d] = o;
    }
}

// ---------------------------------------------------------------------------
// Host-side orchestration
// ---------------------------------------------------------------------------
extern "C" void kernel_launch(void* const* d_in, const int* in_sizes, int n_in,
                              void* d_out, int out_size, void* d_ws, size_t ws_size,
                              hipStream_t stream) {
    (void)in_sizes; (void)n_in; (void)out_size; (void)ws_size;

    const int Bv = 64, Nn = 1024, DIN = 64, DH = 128, Hm = 128, Lnum = 3;
    const size_t NN = (size_t)Nn * Nn;
    const long long ROWS = (long long)Bv * Nn;          // 65536

    const float* x    = (const float*)d_in[0];
    const float* U    = (const float*)d_in[1];
    const float* w1_0 = (const float*)d_in[2];
    const float* b1_0 = (const float*)d_in[3];
    const float* w2_0 = (const float*)d_in[4];
    const float* b2_0 = (const float*)d_in[5];
    const float* w1_r = (const float*)d_in[6];
    const float* b1_r = (const float*)d_in[7];
    const float* w2_r = (const float*)d_in[8];
    const float* b2_r = (const float*)d_in[9];
    const float* bw   = (const float*)d_in[10];
    const float* Wc1  = (const float*)d_in[11];
    const float* bc1  = (const float*)d_in[12];
    const float* alpha= (const float*)d_in[13];
    const float* Wc2  = (const float*)d_in[14];
    const float* bc2  = (const float*)d_in[15];

    // Workspace carve (256B aligned slots)
    size_t off = 0;
    auto carve = [&](size_t bytes) -> char* {
        char* p = (char*)d_ws + off;
        off += (bytes + 255) & ~(size_t)255;
        return p;
    };
    u16*   Ubf   = (u16*)carve(3 * NN * 2);
    u16*   UTbf  = (u16*)carve(3 * NN * 2);
    u16*   hbf   = (u16*)carve((size_t)ROWS * DH * 2);
    u16*   aggbf = (u16*)carve((size_t)ROWS * DH * 2);
    u16*   m1bf  = (u16*)carve((size_t)ROWS * Hm * 2);
    u16*   m2bf  = (u16*)carve((size_t)ROWS * DH * 2);
    float* accum = (float*)carve((size_t)ROWS * DH * 4);
    u16*   w10bf = (u16*)carve((size_t)3 * DIN * Hm * 2);
    u16*   w20bf = (u16*)carve((size_t)3 * Hm * DH * 2);
    u16*   w1rbf = (u16*)carve((size_t)2 * 3 * DH * Hm * 2);
    u16*   w2rbf = (u16*)carve((size_t)2 * 3 * Hm * DH * 2);
    float* wsb   = (float*)carve(16 * 4);

    // --- one-time conversions (cheap; L2-resident thereafter) ---
    k_conv_bf16<<<dim3(2048), dim3(256), 0, stream>>>(x, hbf, ROWS * DIN);
    k_convT<<<dim3(Nn / 32, Nn / 32, 3), dim3(32, 8), 0, stream>>>(U, Ubf, UTbf, Nn);
    k_conv_bf16<<<dim3(96), dim3(256), 0, stream>>>(w1_0, w10bf, (long long)3 * DIN * Hm);
    k_conv_bf16<<<dim3(192), dim3(256), 0, stream>>>(w2_0, w20bf, (long long)3 * Hm * DH);
    k_conv_bf16<<<dim3(384), dim3(256), 0, stream>>>(w1_r, w1rbf, (long long)2 * 3 * DH * Hm);
    k_conv_bf16<<<dim3(384), dim3(256), 0, stream>>>(w2_r, w2rbf, (long long)2 * 3 * Hm * DH);
    k_softmax3<<<dim3(1), dim3(32), 0, stream>>>(bw, wsb, Lnum);

    // --- GCN layers ---
    for (int l = 0; l < Lnum; ++l) {
        const int D = (l == 0) ? DIN : DH;          // input feature width
        k_zero_f32<<<dim3(2048), dim3(256), 0, stream>>>(accum, ROWS * DH);

        for (int k = 0; k < 3; ++k) {
            const u16* w1p = (l == 0) ? (w10bf + (size_t)k * DIN * Hm)
                                      : (w1rbf + ((size_t)(l - 1) * 3 + k) * DH * Hm);
            const u16* w2p = (l == 0) ? (w20bf + (size_t)k * Hm * DH)
                                      : (w2rbf + ((size_t)(l - 1) * 3 + k) * Hm * DH);
            const float* b1p = (l == 0) ? (b1_0 + (size_t)k * Hm)
                                        : (b1_r + ((size_t)(l - 1) * 3 + k) * Hm);
            const float* b2p = (l == 0) ? (b2_0 + (size_t)k * DH)
                                        : (b2_r + ((size_t)(l - 1) * 3 + k) * DH);

            // agg[b] = U[k]^T @ h[b]   -> bf16   (M=1024, N=D, K=1024, batched 64)
            gemm_bf16_wmma<<<dim3(Nn / BM, (D + BN - 1) / BN, Bv), dim3(256),
                             GEMM_LDS_BYTES, stream>>>(
                UTbf + (size_t)k * NN, 0, Nn,
                hbf, (long long)Nn * D, D,
                aggbf, (long long)Nn * D, D,
                nullptr, nullptr, Nn, D, Nn, EP_BF16);

            // m1 = relu(agg @ w1 + b1) -> bf16   (M=65536, N=128, K=D)
            gemm_bf16_wmma<<<dim3(ROWS / BM, 1, 1), dim3(256),
                             GEMM_LDS_BYTES, stream>>>(
                aggbf, 0, D,
                w1p, 0, Hm,
                m1bf, 0, Hm,
                b1p, nullptr, (int)ROWS, Hm, D, EP_BF16_BIAS_RELU);

            // m2 = (m1 @ w2 + b2) * softmax(w3)[k] -> bf16  (M=65536, N=128, K=128)
            gemm_bf16_wmma<<<dim3(ROWS / BM, 1, 1), dim3(256),
                             GEMM_LDS_BYTES, stream>>>(
                m1bf, 0, Hm,
                w2p, 0, DH,
                m2bf, 0, DH,
                b2p, wsb + (size_t)l * 3 + k, (int)ROWS, DH, Hm, EP_BF16_BIAS_SCALE);

            // accum[b] += U[k] @ m2[b]  (f32 accumulate; launches serialize on stream)
            gemm_bf16_wmma<<<dim3(Nn / BM, DH / BN, Bv), dim3(256),
                             GEMM_LDS_BYTES, stream>>>(
                Ubf + (size_t)k * NN, 0, Nn,
                m2bf, (long long)Nn * DH, DH,
                accum, (long long)Nn * DH, DH,
                nullptr, nullptr, Nn, DH, Nn, EP_F32_ACC);
        }

        if (l < Lnum - 1) {
            // h = relu(accum) -> bf16 for next layer
            k_relu_conv_bf16<<<dim3(2048), dim3(256), 0, stream>>>(accum, hbf, ROWS * DH);
        }
    }

    // mean-pool (with final relu) + classifier head -> d_out [64,4] f32
    k_pool_head<<<dim3(Bv), dim3(128), 0, stream>>>(accum, Wc1, bc1, alpha, Wc2, bc2,
                                                    (float*)d_out);
}
